// self_attention_68204080660715
// MI455X (gfx1250) — matrix operand order repaired and checked by reading, not compiled, not run on hardware
//
#include <hip/hip_runtime.h>
#include <hip/hip_bf16.h>

typedef __attribute__((ext_vector_type(16))) _Float16 v16h;
typedef __attribute__((ext_vector_type(8)))  _Float16 v8h;
typedef __attribute__((ext_vector_type(8)))  float    v8f;
typedef __attribute__((ext_vector_type(4)))  float    v4f;

static __device__ __forceinline__ v8f vzero8() {
    v8f z;
    #pragma unroll
    for (int i = 0; i < 8; ++i) z[i] = 0.0f;
    return z;
}

static __device__ __forceinline__ v16h cat16(v8h lo, v8h hi) {
    return __builtin_shufflevector(lo, hi, 0, 1, 2, 3, 4, 5, 6, 7,
                                           8, 9, 10, 11, 12, 13, 14, 15);
}

// Async global -> LDS copy of 16 bytes per lane (no VGPR round-trip, ASYNCcnt).
static __device__ __forceinline__ void async_copy16(const _Float16* g, _Float16* l) {
    const unsigned lds_off = (unsigned)(size_t)l;   // low 32 bits = LDS offset
    asm volatile("global_load_async_to_lds_b128 %0, %1, off"
                 :: "v"(lds_off), "v"(g) : "memory");
}

// -----------------------------------------------------------------------------
// Kernel 0: one-time weight convert+transpose: w[mat,256,64] f32 -> wT[mat,64,256] f16
// -----------------------------------------------------------------------------
__global__ __launch_bounds__(64) void wt_kernel(const float* __restrict__ w,
                                                _Float16* __restrict__ wT) {
    const int mat = blockIdx.x;          // 0..2
    const int e   = threadIdx.x;         // 0..63
    const float* src = w + (size_t)mat * 256 * 64;
    _Float16*    dst = wT + (size_t)mat * 64 * 256 + (size_t)e * 256;
    #pragma unroll 8
    for (int d = 0; d < 256; ++d) dst[d] = (_Float16)src[(size_t)d * 64 + e];
}

// -----------------------------------------------------------------------------
// Kernel 1: QKV projection. x[b,s,256] f32 @ wT -> Q (f16, pre-scaled 1/8),
// K (f16, row-major), Vt (f16, transposed [b,64,s]).
// One wave computes a 16x64 tile: 32x v_wmma_f32_16x16x32_f16, all-b128 feeds.
// -----------------------------------------------------------------------------
__global__ __launch_bounds__(32) void qkv_kernel(const float* __restrict__ x,
                                                 const _Float16* __restrict__ wT,
                                                 _Float16* __restrict__ Q,
                                                 _Float16* __restrict__ K,
                                                 _Float16* __restrict__ Vt) {
    const int lane = threadIdx.x;
    const int col  = lane & 15;
    const int hi   = lane >> 4;
    const int s0   = blockIdx.x * 16;
    const int b    = blockIdx.y;
    const int mat  = blockIdx.z;        // 0=Q 1=K 2=V

    const float*    xrow = x  + ((size_t)b * 4096 + s0 + col) * 256;
    const _Float16* wtm  = wT + (size_t)mat * 64 * 256;

    v8f acc[4] = {vzero8(), vzero8(), vzero8(), vzero8()};

    #pragma unroll
    for (int kc = 0; kc < 8; ++kc) {
        const int d0 = kc * 32;
        v16h a;
        {
            const v4f* g0 = (const v4f*)(xrow + d0 + hi * 8);
            const v4f* g1 = (const v4f*)(xrow + d0 + 16 + hi * 8);
            const v4f x0 = g0[0], x1 = g0[1], x2 = g1[0], x3 = g1[1];
            #pragma unroll
            for (int i = 0; i < 4; ++i) {
                a[i]      = (_Float16)x0[i];
                a[4 + i]  = (_Float16)x1[i];
                a[8 + i]  = (_Float16)x2[i];
                a[12 + i] = (_Float16)x3[i];
            }
        }
        #pragma unroll
        for (int nc = 0; nc < 4; ++nc) {
            const v8h* wp = (const v8h*)(wtm + (size_t)(nc * 16 + col) * 256
                                             + d0 + hi * 8);
            const v16h bm = cat16(wp[0], wp[2]);
            acc[nc] = __builtin_amdgcn_wmma_f32_16x16x32_f16(
                false, a, false, bm, (short)0, acc[nc], false, false);
        }
    }

    const float scale = (mat == 0) ? 0.125f : 1.0f;   // fold 1/sqrt(64) into Q
    if (mat < 2) {
        _Float16* dst = (mat == 0) ? Q : K;
        #pragma unroll
        for (int nc = 0; nc < 4; ++nc)
            #pragma unroll
            for (int r = 0; r < 8; ++r) {
                const int srow = s0 + r + hi * 8;
                dst[((size_t)b * 4096 + srow) * 64 + nc * 16 + col] =
                    (_Float16)(acc[nc][r] * scale);
            }
    } else {
        #pragma unroll
        for (int nc = 0; nc < 4; ++nc)
            #pragma unroll
            for (int r = 0; r < 8; ++r) {
                const int srow = s0 + r + hi * 8;
                const int e    = nc * 16 + col;
                Vt[((size_t)b * 64 + e) * 4096 + srow] = (_Float16)acc[nc][r];
            }
    }
}

// -----------------------------------------------------------------------------
// Kernel 2: fused flash attention, async double-buffered K/V staging.
// Block = 256 threads = 8 waves = 4 q-tiles x 2 key-splits (2048 keys each).
// Iteration pipeline: issue async LDS copies for tile it+1 (buffer ^1),
// s_wait_asynccnt for tile it, barrier, then per-wave: 4 score WMMAs, online
// softmax, P C->A relayout via LDS, 4 PV WMMAs. Log-sum-exp merge at the end.
// -----------------------------------------------------------------------------
__global__ __launch_bounds__(256) void attn_kernel(const _Float16* __restrict__ Q,
                                                   const _Float16* __restrict__ K,
                                                   const _Float16* __restrict__ Vt,
                                                   float* __restrict__ out) {
    // padded rows: 72 (gcd(36,64)=4) / 40 (gcd(20,64)=4) -> conflict-free b128 ds reads
    __shared__ __align__(16) _Float16 ldsK[2][2][32][72];  // [buf][grp][key][e]
    __shared__ __align__(16) _Float16 ldsV[2][2][64][40];  // [buf][grp][e][key]
    __shared__ __align__(16) _Float16 lds_p[8][512];       // per-wave P staging
    __shared__ float lds_m[4][2][16];                      // [qt][grp][row]
    __shared__ float lds_l[4][2][16];
    __shared__ float lds_acc[4][2][16][64];                // partial O per key-split

    const int tid  = threadIdx.x;
    const int wv   = tid >> 5;          // 0..7
    const int qt   = wv & 3;            // q-tile within block
    const int grp  = wv >> 2;           // key-split group 0/1
    const int lane = tid & 31;
    const int col  = lane & 15;
    const int hi   = lane >> 4;
    const int q0   = blockIdx.x * 64 + qt * 16;
    const int b    = blockIdx.y;

    // staging assignments (one K unit + one V unit per thread per group)
    const int krow = tid >> 3, kch = tid & 7;   // 32 rows x 8 x 16B
    const int ve   = tid >> 2, vch = tid & 3;   // 64 rows x 4 x 16B

    const _Float16* kbase = K  + (size_t)b * 4096 * 64;
    const _Float16* vbase = Vt + (size_t)b * 64 * 4096;

    // Q A-tiles, resident across the whole key loop
    v16h aq0, aq1;
    {
        const _Float16* qrow = Q + ((size_t)b * 4096 + q0 + col) * 64;
        const v8h* p0 = (const v8h*)(qrow + hi * 16);
        const v8h* p1 = (const v8h*)(qrow + 8 + hi * 16);
        const v8h* p2 = (const v8h*)(qrow + 32 + hi * 16);
        const v8h* p3 = (const v8h*)(qrow + 40 + hi * 16);
        aq0 = cat16(p0[0], p1[0]);
        aq1 = cat16(p2[0], p3[0]);
    }

    float m[8], l[8];
    #pragma unroll
    for (int r = 0; r < 8; ++r) { m[r] = -1e30f; l[r] = 0.0f; }
    v8f acc[4] = {vzero8(), vzero8(), vzero8(), vzero8()};

    // prologue: async-stage tile 0 into buffer 0
    #pragma unroll
    for (int g = 0; g < 2; ++g) {
        const int kg = g * 2048;
        async_copy16(kbase + (size_t)(kg + krow) * 64 + kch * 8,
                     &ldsK[0][g][krow][kch * 8]);
        async_copy16(vbase + (size_t)ve * 4096 + kg + vch * 8,
                     &ldsV[0][g][ve][vch * 8]);
    }

    for (int it = 0; it < 64; ++it) {
        const int buf = it & 1;
        if (it + 1 < 64) {
            // issue next tile's async copies into the other buffer
            #pragma unroll
            for (int g = 0; g < 2; ++g) {
                const int kg = g * 2048 + (it + 1) * 32;
                async_copy16(kbase + (size_t)(kg + krow) * 64 + kch * 8,
                             &ldsK[buf ^ 1][g][krow][kch * 8]);
                async_copy16(vbase + (size_t)ve * 4096 + kg + vch * 8,
                             &ldsV[buf ^ 1][g][ve][vch * 8]);
            }
            // wait for the oldest 4 async ops (current tile); newest 4 in flight
            asm volatile("s_wait_asynccnt 0x4" ::: "memory");
        } else {
            asm volatile("s_wait_asynccnt 0x0" ::: "memory");
        }
        __syncthreads();

        // ---- scores: two 16(q) x 16(key) blocks over this group's 32 keys ----
        v8f sa = vzero8(), sb = vzero8();
        {
            const v8h* kr0 = (const v8h*)&ldsK[buf][grp][col][hi * 16];
            const v8h* kr1 = (const v8h*)&ldsK[buf][grp][col][32 + hi * 16];
            const v16h b0 = cat16(kr0[0], kr0[1]);   // e rows 0..31
            const v16h b1 = cat16(kr1[0], kr1[1]);   // e rows 32..63
            sa = __builtin_amdgcn_wmma_f32_16x16x32_f16(false, aq0, false, b0,
                                                        (short)0, sa, false, false);
            sa = __builtin_amdgcn_wmma_f32_16x16x32_f16(false, aq1, false, b1,
                                                        (short)0, sa, false, false);
        }
        {
            const v8h* kr0 = (const v8h*)&ldsK[buf][grp][16 + col][hi * 16];
            const v8h* kr1 = (const v8h*)&ldsK[buf][grp][16 + col][32 + hi * 16];
            const v16h b0 = cat16(kr0[0], kr0[1]);
            const v16h b1 = cat16(kr1[0], kr1[1]);
            sb = __builtin_amdgcn_wmma_f32_16x16x32_f16(false, aq0, false, b0,
                                                        (short)0, sb, false, false);
            sb = __builtin_amdgcn_wmma_f32_16x16x32_f16(false, aq1, false, b1,
                                                        (short)0, sb, false, false);
        }

        // ---- online softmax: each C row lives across one 16-lane half ----
        float alpha[8];
        #pragma unroll
        for (int r = 0; r < 8; ++r) {
            float v = fmaxf(sa[r], sb[r]);
            #pragma unroll
            for (int off = 8; off; off >>= 1)
                v = fmaxf(v, __shfl_xor(v, off, 16));
            const float mn = fmaxf(m[r], v);
            alpha[r] = __expf(m[r] - mn);
            m[r] = mn;
        }
        #pragma unroll
        for (int r = 0; r < 8; ++r) {
            const float pa = __expf(sa[r] - m[r]);
            const float pb = __expf(sb[r] - m[r]);
            float rs = pa + pb;
            #pragma unroll
            for (int off = 8; off; off >>= 1)
                rs += __shfl_xor(rs, off, 16);
            l[r] = l[r] * alpha[r] + rs;
            lds_p[wv][(r + hi * 8) * 32 + col]      = (_Float16)pa;
            lds_p[wv][(r + hi * 8) * 32 + 16 + col] = (_Float16)pb;
        }
        __syncthreads();

        // ---- read P back in A layout; rescale; accumulate P @ V ----
        v16h pA;
        {
            const v8h* pr = (const v8h*)&lds_p[wv][col * 32];
            pA = cat16(pr[2 * hi], pr[2 * hi + 1]);
        }
        #pragma unroll
        for (int nc = 0; nc < 4; ++nc) {
            #pragma unroll
            for (int r = 0; r < 8; ++r) acc[nc][r] *= alpha[r];
            const v8h* vr = (const v8h*)&ldsV[buf][grp][nc * 16 + col][hi * 16];
            const v16h bv = cat16(vr[0], vr[1]);
            acc[nc] = __builtin_amdgcn_wmma_f32_16x16x32_f16(
                false, pA, false, bv, (short)0, acc[nc], false, false);
        }
        __syncthreads();   // all reads done before next iteration's async refill
    }

    // ---- per-wave state to LDS ----
    if (col == 0) {
        #pragma unroll
        for (int r = 0; r < 8; ++r) {
            lds_m[qt][grp][r + hi * 8] = m[r];
            lds_l[qt][grp][r + hi * 8] = l[r];
        }
    }
    #pragma unroll
    for (int nc = 0; nc < 4; ++nc)
        #pragma unroll
        for (int r = 0; r < 8; ++r)
            lds_acc[qt][grp][r + hi * 8][nc * 16 + col] = acc[nc][r];
    __syncthreads();

    // ---- log-sum-exp merge of the 2 key-splits; coalesced f32 stores ----
    for (int idx = tid; idx < 4 * 16 * 64; idx += 256) {
        const int mq  = idx >> 10;
        const int row = (idx >> 6) & 15;
        const int e   = idx & 63;
        const float m0 = lds_m[mq][0][row], m1 = lds_m[mq][1][row];
        const float M  = fmaxf(m0, m1);
        const float a0 = __expf(m0 - M), a1 = __expf(m1 - M);
        const float num = a0 * lds_acc[mq][0][row][e] + a1 * lds_acc[mq][1][row][e];
        const float den = a0 * lds_l[mq][0][row] + a1 * lds_l[mq][1][row];
        out[((size_t)b * 4096 + blockIdx.x * 64 + mq * 16 + row) * 64 + e] = num / den;
    }
}

extern "C" void kernel_launch(void* const* d_in, const int* in_sizes, int n_in,
                              void* d_out, int out_size, void* d_ws, size_t ws_size,
                              hipStream_t stream) {
    (void)in_sizes; (void)n_in; (void)out_size; (void)ws_size;
    const float* x = (const float*)d_in[0];   // [4,4096,256] f32
    const float* w = (const float*)d_in[1];   // [3,256,64]  f32
    float* out = (float*)d_out;               // [4,4096,64] f32

    _Float16* Q  = (_Float16*)d_ws;                       // 2 MB
    _Float16* K  = Q  + (size_t)4 * 4096 * 64;            // 2 MB
    _Float16* Vt = K  + (size_t)4 * 4096 * 64;            // 2 MB (transposed V)
    _Float16* wT = Vt + (size_t)4 * 64 * 4096;            // 96 KB (f16, transposed)

    wt_kernel<<<dim3(3), 64, 0, stream>>>(w, wT);
    qkv_kernel<<<dim3(256, 4, 3), 32, 0, stream>>>(x, wT, Q, K, Vt);
    attn_kernel<<<dim3(64, 4), 256, 0, stream>>>(Q, K, Vt, out);
}
